// GCSAN_77996606095969
// MI455X (gfx1250) — matrix operand order repaired and checked by reading, not compiled, not run on hardware
//
#include <hip/hip_runtime.h>
#include <math.h>

#define WAVES 4
constexpr int Bn  = 256;
constexpr int Ln  = 200;
constexpr int Dn  = 128;
constexpr int NHn = 2;
constexpr int HDn = 64;
constexpr int BLn = Bn * Ln;
constexpr float LAMBn = 0.6f;

typedef __attribute__((ext_vector_type(16))) __bf16        v16bf;
typedef __attribute__((ext_vector_type(8)))  float         v8f;
typedef __attribute__((ext_vector_type(4)))  float         f32x4;
typedef __attribute__((ext_vector_type(4)))  unsigned int  u32x4;
typedef __attribute__((ext_vector_type(8)))  unsigned int  u32x8;

__device__ __forceinline__ unsigned short f32_bf16_bits(float f) {
  __bf16 h = (__bf16)f;                       // RNE
  return __builtin_bit_cast(unsigned short, h);
}
__device__ __forceinline__ unsigned int pack2(float a, float b) {
  return (unsigned int)f32_bf16_bits(a) | ((unsigned int)f32_bf16_bits(b) << 16);
}
// fragment = two aligned 16B LDS loads -> 32B -> v16bf
__device__ __forceinline__ v16bf frag_ld(const unsigned int* p, const unsigned int* q) {
  u32x4 a = *(const u32x4*)p;
  u32x4 c = *(const u32x4*)q;
  u32x8 u = __builtin_shufflevector(a, c, 0, 1, 2, 3, 4, 5, 6, 7);
  return __builtin_bit_cast(v16bf, u);
}
// interior row: 8 x b128 loads at immediate offsets -> 4 x b128 LDS stores
__device__ __forceinline__ void stage_fast(const float* __restrict__ p,
                                           unsigned int* __restrict__ dst32) {
  unsigned int pk[16];
  #pragma unroll
  for (int j = 0; j < 8; ++j) {
    f32x4 v4 = *(const f32x4*)(p + 4 * j);
    pk[2 * j]     = pack2(v4.x, v4.y);
    pk[2 * j + 1] = pack2(v4.z, v4.w);
  }
  u32x4* dst = (u32x4*)dst32;
  dst[0] = u32x4{pk[0],  pk[1],  pk[2],  pk[3]};
  dst[1] = u32x4{pk[4],  pk[5],  pk[6],  pk[7]};
  dst[2] = u32x4{pk[8],  pk[9],  pk[10], pk[11]};
  dst[3] = u32x4{pk[12], pk[13], pk[14], pk[15]};
}
// K-tail round (taken once, only when K%64 != 0): zero-fill k >= K.
// prow = row base (clamped, always addressable); kloc0 = this lane's 32-half.
__device__ __forceinline__ void stage_tail(const float* __restrict__ prow, int kmain,
                                           int kloc0, int rem,
                                           unsigned int* __restrict__ dst32) {
  unsigned int pk[16];
  #pragma unroll
  for (int j = 0; j < 8; ++j) {
    int kloc = kloc0 + 4 * j;
    bool ok = kloc < rem;                       // rem%4==0 => all-or-nothing
    const float* pj = ok ? (prow + kmain + kloc) : prow;
    f32x4 v4 = *(const f32x4*)pj;
    f32x4 zz = {0.f, 0.f, 0.f, 0.f};
    v4 = ok ? v4 : zz;
    pk[2 * j]     = pack2(v4.x, v4.y);
    pk[2 * j + 1] = pack2(v4.z, v4.w);
  }
  u32x4* dst = (u32x4*)dst32;
  dst[0] = u32x4{pk[0],  pk[1],  pk[2],  pk[3]};
  dst[1] = u32x4{pk[4],  pk[5],  pk[6],  pk[7]};
  dst[2] = u32x4{pk[8],  pk[9],  pk[10], pk[11]};
  dst[3] = u32x4{pk[12], pk[13], pk[14], pk[15]};
}
// two WMMA sub-steps over a staged 64-deep tile pair
__device__ __forceinline__ v8f wmma2(const unsigned int* sAw, const unsigned int* sBw,
                                     int lm, int g, v8f acc) {
  #pragma unroll
  for (int s = 0; s < 2; ++s) {
    v16bf af = frag_ld(&sAw[lm * 36 + 16 * s + g * 4],
                       &sAw[lm * 36 + 16 * s + 8 + g * 4]);
    v16bf bf = frag_ld(&sBw[lm * 36 + 16 * s + g * 4],
                       &sBw[lm * 36 + 16 * s + 8 + g * 4]);
    acc = __builtin_amdgcn_wmma_f32_16x16x32_bf16(false, af, false, bf,
                                                  (short)0, acc, false, false);
  }
  return acc;
}

// ---------------------------------------------------------------------------
// Generic WMMA GEMM:  C[m,n] = alpha*sum_k Aop[m,k]*Bop[k,n] (+bias)(+res)(relu)
// A row-major (M x K). B: transB ? (N x K) row-major : (K x N) row-major.
// Two-level batch: z -> (b=z/nh, h=z%nh).  One wave = one 16x16 C tile,
// K-step 64 = 2 x v_wmma_f32_16x16x32_bf16 per staging round.
// M/N edge rows are loaded clamped (garbage tolerated, masked at the store);
// only the K tail (once, K%64!=0) zero-fills.  Main loop is guard-free.
// ---------------------------------------------------------------------------
__global__ __launch_bounds__(128)
void gemm_wmma(const float* __restrict__ A, int lda, long sAb, long sAh,
               const float* __restrict__ Bm, int ldb, long sBb, long sBh,
               float* __restrict__ C, int ldc, long sCb, long sCh,
               const float* __restrict__ bias,
               const float* __restrict__ res, int ldr,
               float alpha, int M, int N, int K, int transB, int relu, int nh)
{
  __shared__ unsigned int sA32[WAVES][16 * 36];
  __shared__ unsigned int sB32[WAVES][16 * 36];
  const int tid  = threadIdx.x;
  const int w    = tid >> 5;
  const int lane = tid & 31;
  const int z = blockIdx.z;
  const int b = z / nh, h = z - b * nh;
  const float* Ab = A  + (long)b * sAb + (long)h * sAh;
  const float* Bb = Bm + (long)b * sBb + (long)h * sBh;
  float*       Cb = C  + (long)b * sCb + (long)h * sCh;
  const int n0 = blockIdx.x * 16;
  const int m0 = (blockIdx.y * WAVES + w) * 16;
  const int g  = lane >> 4;
  const int lm = lane & 15;
  unsigned int*   sAw   = sA32[w];
  unsigned int*   sBw   = sB32[w];
  unsigned short* sBw16 = (unsigned short*)sBw;

  const int rrow  = lane >> 1;   // staging row (0..15)
  const int rhalf = lane & 1;    // which 32-wide K half

  // clamped row base pointers (always addressable; edge rows masked at store)
  const int mA = m0 + rrow, nB = n0 + rrow;
  const float* pAbase = Ab + (long)(mA < M ? mA : 0) * lda;
  const float* pBbase = Bb + (long)(nB < N ? nB : 0) * ldb;
  const float* pArow  = pAbase + rhalf * 32;
  const float* pBrow  = pBbase + rhalf * 32;
  unsigned int* dstA = &sAw[rrow * 36 + rhalf * 16];
  unsigned int* dstB = &sBw[rrow * 36 + rhalf * 16];

  v8f acc = {0.f, 0.f, 0.f, 0.f, 0.f, 0.f, 0.f, 0.f};

  const int kmain = K & ~63;
  for (int k0 = 0; k0 < kmain; k0 += 64) {       // guard-free steady state
    stage_fast(pArow + k0, dstA);
    if (transB) {
      stage_fast(pBrow + k0, dstB);
    } else {
      #pragma unroll
      for (int rep = 0; rep < 2; ++rep) {
        int koff = lane + 32 * rep;              // 0..63, all rows < kmain <= K
        const float* pr = Bb + (long)(k0 + koff) * ldb + n0;
        #pragma unroll
        for (int j = 0; j < 4; ++j) {
          f32x4 v4 = *(const f32x4*)(pr + 4 * j);
          sBw16[(4 * j + 0) * 72 + koff] = f32_bf16_bits(v4.x);
          sBw16[(4 * j + 1) * 72 + koff] = f32_bf16_bits(v4.y);
          sBw16[(4 * j + 2) * 72 + koff] = f32_bf16_bits(v4.z);
          sBw16[(4 * j + 3) * 72 + koff] = f32_bf16_bits(v4.w);
        }
      }
    }
    __syncthreads();
    acc = wmma2(sAw, sBw, lm, g, acc);
    __syncthreads();
  }
  if (kmain < K) {                               // single tail round (K=200)
    const int rem = K - kmain;                   // multiple of 4
    stage_tail(pAbase, kmain, rhalf * 32, rem, dstA);
    if (transB) {
      stage_tail(pBbase, kmain, rhalf * 32, rem, dstB);
    } else {
      #pragma unroll
      for (int rep = 0; rep < 2; ++rep) {
        int koff = lane + 32 * rep;
        bool ok = koff < rem;
        long ro = (long)(ok ? kmain + koff : 0) * ldb;
        #pragma unroll
        for (int j = 0; j < 4; ++j) {
          f32x4 v4 = *(const f32x4*)(Bb + ro + n0 + 4 * j);
          f32x4 zz = {0.f, 0.f, 0.f, 0.f};
          v4 = ok ? v4 : zz;
          sBw16[(4 * j + 0) * 72 + koff] = f32_bf16_bits(v4.x);
          sBw16[(4 * j + 1) * 72 + koff] = f32_bf16_bits(v4.y);
          sBw16[(4 * j + 2) * 72 + koff] = f32_bf16_bits(v4.z);
          sBw16[(4 * j + 3) * 72 + koff] = f32_bf16_bits(v4.w);
        }
      }
    }
    __syncthreads();
    acc = wmma2(sAw, sBw, lm, g, acc);
    __syncthreads();
  }

  #pragma unroll
  for (int r = 0; r < 8; ++r) {
    int mm = m0 + g * 8 + r;
    int nn = n0 + lm;
    if (mm < M && nn < N) {
      float val = acc[r] * alpha;
      if (bias) val += bias[nn];
      if (res)  val += res[(long)mm * ldr + nn];
      if (relu) val = val > 0.f ? val : 0.f;
      Cb[(long)mm * ldc + nn] = val;
    }
  }
}

// ---------------------------------------------------------------------------
// Graph build: one block per row b.
// ---------------------------------------------------------------------------
__global__ __launch_bounds__(256)
void build_graph(const int* __restrict__ seqs,
                 int* __restrict__ items_g, int* __restrict__ alias_g,
                 float* __restrict__ Ain, float* __restrict__ Aout)
{
  __shared__ int s[Ln];
  __shared__ int items[Ln];
  __shared__ int aliasS[Ln];
  __shared__ unsigned int bits[Ln * 7];
  __shared__ float indeg[Ln], outdeg[Ln];
  __shared__ int nu_s;
  const int b = blockIdx.x;
  const int t = threadIdx.x;
  const int* seq = seqs + b * Ln;

  if (t < Ln) s[t] = seq[t];
  __syncthreads();
  for (int it = 0; it < Ln; ++it) {       // odd-even transposition sort
    int i = 2 * t + (it & 1);
    if (i + 1 < Ln) {
      int a0 = s[i], a1 = s[i + 1];
      if (a0 > a1) { s[i] = a1; s[i + 1] = a0; }
    }
    __syncthreads();
  }
  if (t == 0) {
    int nu = 0;
    for (int i = 0; i < Ln; ++i)
      if (i == 0 || s[i] != s[i - 1]) items[nu++] = s[i];
    for (int i = nu; i < Ln; ++i) items[i] = 0;
    nu_s = nu;
  }
  for (int i = t; i < Ln * 7; i += 256) bits[i] = 0u;
  __syncthreads();
  const int nu = nu_s;
  if (t < Ln) {
    int v = seq[t];
    int lo = 0, hi = nu;
    while (lo < hi) { int mid = (lo + hi) >> 1; if (items[mid] < v) lo = mid + 1; else hi = mid; }
    aliasS[t] = lo;
    alias_g[b * Ln + t] = lo;
    items_g[b * Ln + t] = items[t];
  }
  __syncthreads();
  if (t < Ln - 1 && seq[t + 1] > 0) {
    int u = aliasS[t], vv = aliasS[t + 1];
    atomicOr(&bits[u * 7 + (vv >> 5)], 1u << (vv & 31));
  }
  __syncthreads();
  if (t < Ln) {
    int od = 0;
    for (int wd = 0; wd < 7; ++wd) od += __popc(bits[t * 7 + wd]);
    outdeg[t] = od > 0 ? (float)od : 1.f;
    int id = 0;
    for (int j = 0; j < Ln; ++j) id += (bits[j * 7 + (t >> 5)] >> (t & 31)) & 1;
    indeg[t] = id > 0 ? (float)id : 1.f;
  }
  __syncthreads();
  float* Ai = Ain  + (long)b * Ln * Ln;
  float* Ao = Aout + (long)b * Ln * Ln;
  for (int idx = t; idx < Ln * Ln; idx += 256) {
    int i = idx / Ln, j = idx - i * Ln;
    float bij = (float)((bits[i * 7 + (j >> 5)] >> (j & 31)) & 1);
    float bji = (float)((bits[j * 7 + (i >> 5)] >> (i & 31)) & 1);
    Ai[idx] = bji / indeg[i];
    Ao[idx] = bij / outdeg[i];
  }
}

__global__ void gather_emb(const float* __restrict__ emb, const int* __restrict__ items,
                           float* __restrict__ hidden)
{
  long i = (long)blockIdx.x * blockDim.x + threadIdx.x;
  if (i < (long)BLn * Dn) {
    long row = i / Dn; int d = (int)(i - row * Dn);
    hidden[i] = emb[(long)items[row] * Dn + d];
  }
}

__global__ void gru_update(const float* __restrict__ gi, const float* __restrict__ gh,
                           float* __restrict__ hidden)
{
  long i = (long)blockIdx.x * blockDim.x + threadIdx.x;
  if (i < (long)BLn * Dn) {
    long m = i / Dn; int d = (int)(i - m * Dn);
    const float* gim = gi + m * (3 * Dn);
    const float* ghm = gh + m * (3 * Dn);
    float ir = gim[d], ii = gim[Dn + d], in_ = gim[2 * Dn + d];
    float hr = ghm[d], hi = ghm[Dn + d], hn  = ghm[2 * Dn + d];
    float r  = 1.f / (1.f + __expf(-(ir + hr)));
    float zz = 1.f / (1.f + __expf(-(ii + hi)));
    float ng = tanhf(in_ + r * hn);
    hidden[i] = ng + zz * (hidden[i] - ng);
  }
}

__global__ void assemble_x(const float* __restrict__ hidden, const int* __restrict__ alias,
                           const float* __restrict__ pos, const int* __restrict__ seqs,
                           const int* __restrict__ lens,
                           float* __restrict__ x, float* __restrict__ ht)
{
  long i = (long)blockIdx.x * blockDim.x + threadIdx.x;
  if (i < (long)BLn * Dn) {
    int d = (int)(i % Dn);
    long bl = i / Dn;
    int b = (int)(bl / Ln), l = (int)(bl - (long)b * Ln);
    int al = alias[b * Ln + l];
    float sh = hidden[((long)b * Ln + al) * Dn + d];
    float msk = (seqs[b * Ln + l] > 0) ? 1.f : 0.f;
    x[i] = (sh + pos[l * Dn + d]) * msk;
    if (l == lens[b] - 1) ht[b * Dn + d] = sh;
  }
}

__global__ __launch_bounds__(128)
void layernorm(const float* __restrict__ in, const float* __restrict__ g,
               const float* __restrict__ bb, float* __restrict__ out)
{
  __shared__ float red[128];
  long row = blockIdx.x;
  int d = threadIdx.x;
  float v = in[row * Dn + d];
  red[d] = v; __syncthreads();
  for (int s = 64; s > 0; s >>= 1) { if (d < s) red[d] += red[d + s]; __syncthreads(); }
  float mu = red[0] / Dn; __syncthreads();
  float df = v - mu;
  red[d] = df * df; __syncthreads();
  for (int s = 64; s > 0; s >>= 1) { if (d < s) red[d] += red[d + s]; __syncthreads(); }
  float var = red[0] / Dn;
  out[row * Dn + d] = df * rsqrtf(var + 1e-5f) * g[d] + bb[d];
}

__global__ __launch_bounds__(256)
void softmax_causal(float* __restrict__ sc)
{
  __shared__ float red[256];
  long row = blockIdx.x;            // (B*NH*L) rows
  int q = (int)(row % Ln);
  float* p = sc + row * Ln;
  int t = threadIdx.x;
  float v = -3.0e38f;
  if (t < Ln) v = (t > q) ? -1.0e9f : p[t];
  red[t] = v; __syncthreads();
  for (int s = 128; s > 0; s >>= 1) { if (t < s) red[t] = fmaxf(red[t], red[t + s]); __syncthreads(); }
  float mx = red[0]; __syncthreads();
  float e = (t < Ln) ? __expf(v - mx) : 0.f;
  red[t] = e; __syncthreads();
  for (int s = 128; s > 0; s >>= 1) { if (t < s) red[t] += red[t + s]; __syncthreads(); }
  float sum = red[0];
  if (t < Ln) p[t] = e / sum;
}

__global__ __launch_bounds__(128)
void final_out(const float* __restrict__ x3, const int* __restrict__ lens,
               const float* __restrict__ ht, const float* __restrict__ g,
               const float* __restrict__ bb, float* __restrict__ out)
{
  __shared__ float red[128];
  int b = blockIdx.x, d = threadIdx.x;
  int l = lens[b] - 1;
  float v = x3[((long)b * Ln + l) * Dn + d];
  red[d] = v; __syncthreads();
  for (int s = 64; s > 0; s >>= 1) { if (d < s) red[d] += red[d + s]; __syncthreads(); }
  float mu = red[0] / Dn; __syncthreads();
  float df = v - mu;
  red[d] = df * df; __syncthreads();
  for (int s = 64; s > 0; s >>= 1) { if (d < s) red[d] += red[d + s]; __syncthreads(); }
  float var = red[0] / Dn;
  float lf = df * rsqrtf(var + 1e-5f) * g[d] + bb[d];
  out[b * Dn + d] = LAMBn * lf + (1.f - LAMBn) * ht[b * Dn + d];
}

// ---------------------------------------------------------------------------
extern "C" void kernel_launch(void* const* d_in, const int* in_sizes, int n_in,
                              void* d_out, int out_size, void* d_ws, size_t ws_size,
                              hipStream_t stream)
{
  const float* item_emb = (const float*)d_in[0];
  const float* pos_emb  = (const float*)d_in[1];
  const float* W_ei = (const float*)d_in[2];  const float* b_ei = (const float*)d_in[3];
  const float* W_eo = (const float*)d_in[4];  const float* b_eo = (const float*)d_in[5];
  const float* b_iah = (const float*)d_in[6]; const float* b_oah = (const float*)d_in[7];
  const float* w_ih = (const float*)d_in[8];  const float* b_ih = (const float*)d_in[9];
  const float* w_hh = (const float*)d_in[10]; const float* b_hh = (const float*)d_in[11];
  const float* ln1_g = (const float*)d_in[12]; const float* ln1_b = (const float*)d_in[13];
  const float* Wq = (const float*)d_in[14]; const float* bq = (const float*)d_in[15];
  const float* Wk = (const float*)d_in[16]; const float* bk = (const float*)d_in[17];
  const float* Wv = (const float*)d_in[18]; const float* bv = (const float*)d_in[19];
  const float* Wo = (const float*)d_in[20]; const float* bo = (const float*)d_in[21];
  const float* ln2_g = (const float*)d_in[22]; const float* ln2_b = (const float*)d_in[23];
  const float* W1 = (const float*)d_in[24]; const float* b1 = (const float*)d_in[25];
  const float* W2 = (const float*)d_in[26]; const float* b2 = (const float*)d_in[27];
  const float* lnf_g = (const float*)d_in[28]; const float* lnf_b = (const float*)d_in[29];
  const int* seqs = (const int*)d_in[30];
  const int* lens = (const int*)d_in[31];
  float* out = (float*)d_out;

  char* ws = (char*)d_ws;
  size_t off = 0;
  auto alloc = [&](size_t bytes) -> void* {
    void* p = ws + off; off += (bytes + 255) & ~(size_t)255; return p;
  };
  int*   items  = (int*)alloc((size_t)BLn * 4);
  int*   alias  = (int*)alloc((size_t)BLn * 4);
  float* hidden = (float*)alloc((size_t)BLn * Dn * 4);
  float* ht     = (float*)alloc((size_t)Bn * Dn * 4);
  char* region = ws + off;
  size_t ro;
  auto ralloc = [&](size_t bytes) -> void* {
    void* p = region + ro; ro += (bytes + 255) & ~(size_t)255; return p;
  };
  // phase 1 (graph + GGNN)
  ro = 0;
  float* Ain  = (float*)ralloc((size_t)Bn * Ln * Ln * 4);
  float* Aout = (float*)ralloc((size_t)Bn * Ln * Ln * 4);
  float* hA   = (float*)ralloc((size_t)BLn * Dn * 4);
  float* hB   = (float*)ralloc((size_t)BLn * Dn * 4);
  float* hcat = (float*)ralloc((size_t)BLn * 2 * Dn * 4);
  float* gi   = (float*)ralloc((size_t)BLn * 3 * Dn * 4);
  float* gh   = (float*)ralloc((size_t)BLn * 3 * Dn * 4);
  // phase 2 (attention) — same region, phase-1 buffers are dead by then
  ro = 0;
  float* x      = (float*)ralloc((size_t)BLn * Dn * 4);
  float* qin    = (float*)ralloc((size_t)BLn * Dn * 4);
  float* q      = (float*)ralloc((size_t)BLn * Dn * 4);
  float* kk     = (float*)ralloc((size_t)BLn * Dn * 4);
  float* v      = (float*)ralloc((size_t)BLn * Dn * 4);
  float* scores = (float*)ralloc((size_t)Bn * NHn * Ln * Ln * 4);
  float* att    = (float*)ralloc((size_t)BLn * Dn * 4);
  float* x2     = (float*)ralloc((size_t)BLn * Dn * 4);
  float* h2     = (float*)ralloc((size_t)BLn * Dn * 4);
  float* f1     = (float*)ralloc((size_t)BLn * Dn * 4);

  auto gemm = [&](const float* A, int lda, long sAb, long sAh,
                  const float* Bm, int ldb, long sBb, long sBh,
                  float* C, int ldc, long sCb, long sCh,
                  const float* bias, const float* res, int ldr,
                  float alpha, int M, int N, int K, int transB, int relu,
                  int nh, int batch) {
    dim3 grid((N + 15) / 16, (M + 16 * WAVES - 1) / (16 * WAVES), batch);
    gemm_wmma<<<grid, 128, 0, stream>>>(A, lda, sAb, sAh, Bm, ldb, sBb, sBh,
                                        C, ldc, sCb, sCh, bias, res, ldr,
                                        alpha, M, N, K, transB, relu, nh);
  };
  const long elemN = (long)BLn * Dn;
  const int eBlocks = (int)((elemN + 255) / 256);

  // ===== Phase 1: graph + GGNN =====
  build_graph<<<Bn, 256, 0, stream>>>(seqs, items, alias, Ain, Aout);
  gather_emb<<<eBlocks, 256, 0, stream>>>(item_emb, items, hidden);
  gemm(hidden, Dn, 0, 0, W_ei, Dn, 0, 0, hA, Dn, 0, 0, b_ei, nullptr, 0,
       1.f, BLn, Dn, Dn, 1, 0, 1, 1);
  gemm(hidden, Dn, 0, 0, W_eo, Dn, 0, 0, hB, Dn, 0, 0, b_eo, nullptr, 0,
       1.f, BLn, Dn, Dn, 1, 0, 1, 1);
  gemm(Ain, Ln, (long)Ln * Ln, 0, hA, Dn, (long)Ln * Dn, 0,
       hcat, 2 * Dn, (long)Ln * 2 * Dn, 0, b_iah, nullptr, 0,
       1.f, Ln, Dn, Ln, 0, 0, 1, Bn);
  gemm(Aout, Ln, (long)Ln * Ln, 0, hB, Dn, (long)Ln * Dn, 0,
       hcat + Dn, 2 * Dn, (long)Ln * 2 * Dn, 0, b_oah, nullptr, 0,
       1.f, Ln, Dn, Ln, 0, 0, 1, Bn);
  gemm(hcat, 2 * Dn, 0, 0, w_ih, 2 * Dn, 0, 0, gi, 3 * Dn, 0, 0, b_ih, nullptr, 0,
       1.f, BLn, 3 * Dn, 2 * Dn, 1, 0, 1, 1);
  gemm(hidden, Dn, 0, 0, w_hh, Dn, 0, 0, gh, 3 * Dn, 0, 0, b_hh, nullptr, 0,
       1.f, BLn, 3 * Dn, Dn, 1, 0, 1, 1);
  gru_update<<<eBlocks, 256, 0, stream>>>(gi, gh, hidden);

  // ===== Phase 2: transformer block =====
  assemble_x<<<eBlocks, 256, 0, stream>>>(hidden, alias, pos_emb, seqs, lens, x, ht);
  layernorm<<<BLn, 128, 0, stream>>>(x, ln1_g, ln1_b, qin);
  gemm(qin, Dn, 0, 0, Wq, Dn, 0, 0, q,  Dn, 0, 0, bq, nullptr, 0, 1.f, BLn, Dn, Dn, 1, 0, 1, 1);
  gemm(x,   Dn, 0, 0, Wk, Dn, 0, 0, kk, Dn, 0, 0, bk, nullptr, 0, 1.f, BLn, Dn, Dn, 1, 0, 1, 1);
  gemm(x,   Dn, 0, 0, Wv, Dn, 0, 0, v,  Dn, 0, 0, bv, nullptr, 0, 1.f, BLn, Dn, Dn, 1, 0, 1, 1);
  gemm(q, Dn, (long)Ln * Dn, HDn, kk, Dn, (long)Ln * Dn, HDn,
       scores, Ln, (long)NHn * Ln * Ln, (long)Ln * Ln, nullptr, nullptr, 0,
       0.125f, Ln, Ln, HDn, 1, 0, NHn, Bn * NHn);
  softmax_causal<<<Bn * NHn * Ln, 256, 0, stream>>>(scores);
  gemm(scores, Ln, (long)NHn * Ln * Ln, (long)Ln * Ln, v, Dn, (long)Ln * Dn, HDn,
       att, Dn, (long)Ln * Dn, HDn, nullptr, nullptr, 0,
       1.f, Ln, HDn, Ln, 0, 0, NHn, Bn * NHn);
  gemm(att, Dn, 0, 0, Wo, Dn, 0, 0, x2, Dn, 0, 0, bo, qin, Dn,
       1.f, BLn, Dn, Dn, 1, 0, 1, 1);
  layernorm<<<BLn, 128, 0, stream>>>(x2, ln2_g, ln2_b, h2);
  gemm(h2, Dn, 0, 0, W1, Dn, 0, 0, f1, Dn, 0, 0, b1, nullptr, 0,
       1.f, BLn, Dn, Dn, 1, 1, 1, 1);
  gemm(f1, Dn, 0, 0, W2, Dn, 0, 0, x, Dn, 0, 0, b2, h2, Dn,   // x reused as x3
       1.f, BLn, Dn, Dn, 1, 0, 1, 1);
  final_out<<<Bn, 128, 0, stream>>>(x, lens, ht, lnf_g, lnf_b, out);
  (void)in_sizes; (void)n_in; (void)out_size; (void)ws_size;
}
